// SelfAttention_68324339745165
// MI455X (gfx1250) — compile-verified, matrix operand order checked by reading, not run on hardware
//
#include <hip/hip_runtime.h>

typedef __attribute__((ext_vector_type(16))) _Float16 v16h;
typedef __attribute__((ext_vector_type(8)))  _Float16 v8h;
typedef __attribute__((ext_vector_type(8)))  float    v8f;
typedef __attribute__((ext_vector_type(4)))  float    v4f;
typedef __attribute__((ext_vector_type(4)))  int      v4i;

#define HIDDEN 256
#define SEQ    1024
#define HEADS  4
#define HDIM   64
#define BATCH  16

#if defined(__has_builtin)
#if __has_builtin(__builtin_amdgcn_global_load_async_to_lds_b128) && \
    __has_builtin(__builtin_amdgcn_s_wait_asynccnt)
#define HAVE_ASYNC_LDS 1
typedef __attribute__((address_space(1))) v4i* gv4i_p;   // global int4*
typedef __attribute__((address_space(3))) v4i* lv4i_p;   // LDS int4*
#endif
#endif

// ---------------------------------------------------------------------------
// 16-bit WMMA A/B fragment helpers (v_wmma_f32_16x16x32_f16 layout):
// element i of the v16h lives at K = 16*(i/8) + 8*(lane/16) + (i%8), i.e. two
// runs of 8 consecutive K values per lane -> two 16-byte loads when the K dim
// is contiguous in memory.
// ---------------------------------------------------------------------------
__device__ __forceinline__ v16h combine8(v8h lo, v8h hiv) {
  v16h r;
#pragma unroll
  for (int i = 0; i < 8; ++i) { r[i] = lo[i]; r[i + 8] = hiv[i]; }
  return r;
}

__device__ __forceinline__ v16h frag_f16(const _Float16* row_k0, int hi) {
  return combine8(*(const v8h*)(row_k0 + 8 * hi),
                  *(const v8h*)(row_k0 + 16 + 8 * hi));
}

__device__ __forceinline__ v16h frag_f32(const float* row_k0, int hi) {
  const v4f a0 = *(const v4f*)(row_k0 + 8 * hi);
  const v4f a1 = *(const v4f*)(row_k0 + 8 * hi + 4);
  const v4f b0 = *(const v4f*)(row_k0 + 16 + 8 * hi);
  const v4f b1 = *(const v4f*)(row_k0 + 16 + 8 * hi + 4);
  v16h r;
#pragma unroll
  for (int i = 0; i < 4; ++i) {
    r[i]      = (_Float16)a0[i];
    r[i + 4]  = (_Float16)a1[i];
    r[i + 8]  = (_Float16)b0[i];
    r[i + 12] = (_Float16)b1[i];
  }
  return r;
}

// ---------------------------------------------------------------------------
// Projection: y = xs @ W^T + b.  One 256-thread block per (16-row s-tile, b):
// stage x tile (256c x 16s) through LDS (async DMA when available), transpose
// + convert to f16 once, then 8 waves each produce two 16x16 WMMA tiles.
// VT==0: store [B,S,C] f16 (Q,K).  VT==1: store V transposed [B,h,d,S] f16.
// ---------------------------------------------------------------------------
template <int VT>
__global__ __launch_bounds__(256)
void proj_kernel(const float* __restrict__ x_cs,   // [B, C, S]
                 const float* __restrict__ W,      // [C, C] (row = out ch)
                 const float* __restrict__ bias,   // [C]
                 _Float16* __restrict__ y)
{
#ifdef HAVE_ASYNC_LDS
  __shared__ __attribute__((aligned(16))) float xa32[HIDDEN][16];   // raw [c][s]
#endif
  __shared__ __attribute__((aligned(16))) _Float16 a16[16][HIDDEN]; // [s][c] f16

  const int tid  = threadIdx.x;
  const int wave = tid >> 5;
  const int lane = tid & 31;
  const int n    = lane & 15;
  const int hi   = lane >> 4;
  const int s0   = blockIdx.x * 16;
  const int b    = blockIdx.z;
  const float* xb = x_cs + (size_t)b * HIDDEN * SEQ;

  // ---- stage x tile: thread t owns channel row c = t (16 contiguous s) ----
  {
    const float* src = xb + (size_t)tid * SEQ + s0;
    float tmp[16];
#ifdef HAVE_ASYNC_LDS
#pragma unroll
    for (int q = 0; q < 4; ++q)
      __builtin_amdgcn_global_load_async_to_lds_b128(
          (gv4i_p)(src + 4 * q), (lv4i_p)(&xa32[tid][4 * q]), 0, 0);
    __builtin_amdgcn_s_wait_asynccnt(0);
#pragma unroll
    for (int j = 0; j < 16; ++j) tmp[j] = xa32[tid][j];
#else
#pragma unroll
    for (int q = 0; q < 4; ++q) {
      const v4f v = *(const v4f*)(src + 4 * q);
#pragma unroll
      for (int j = 0; j < 4; ++j) tmp[4 * q + j] = v[j];
    }
#endif
#pragma unroll
    for (int j = 0; j < 16; ++j) a16[j][tid] = (_Float16)tmp[j];  // transpose
  }
  __syncthreads();

  // ---- each wave: two 16x16 N-tiles over K=256 ----
#pragma unroll
  for (int t = 0; t < 2; ++t) {
    const int n0 = (wave * 2 + t) * 16;
    const float* wrow = W + (size_t)(n0 + n) * HIDDEN;
    v8f acc = {};
    for (int k0 = 0; k0 < HIDDEN; k0 += 32) {
      const v16h af = frag_f16(&a16[n][k0], hi);   // ds b128
      const v16h bf = frag_f32(wrow + k0, hi);     // glb b128
      acc = __builtin_amdgcn_wmma_f32_16x16x32_f16(false, af, false, bf,
                                                   (short)0, acc, false, false);
    }
    const float bv = bias[n0 + n];
    if (VT == 0) {
      _Float16* yb = y + (size_t)b * SEQ * HIDDEN;
#pragma unroll
      for (int r = 0; r < 8; ++r)
        yb[(size_t)(s0 + r + 8 * hi) * HIDDEN + (n0 + n)] = (_Float16)(acc[r] + bv);
    } else {
      const int c = n0 + n, hh = c >> 6, dd = c & 63;
      v8h pk;
#pragma unroll
      for (int r = 0; r < 8; ++r) pk[r] = (_Float16)(acc[r] + bv);
      *(v8h*)(y + ((size_t)(b * HEADS + hh) * HDIM + dd) * SEQ + s0 + 8 * hi) = pk;
    }
  }
}

// ---------------------------------------------------------------------------
// Attention for one (batch, head, 16-query tile), 256 threads.
//   A) scores = QK^T * 1/8 -> 16x1024 f32 in LDS (register-double-buffered)
//   B) register softmax; probs re-written f16 in-place over the LDS block
//   B2) coalesced non-temporal stream-out of attn (256 B/thread, v4f)
//   C) ctx = P @ V: 8 waves = 4 d-slices x 2 K-halves, LDS reduction
// ---------------------------------------------------------------------------
__global__ __launch_bounds__(256)
void attention_kernel(const _Float16* __restrict__ qm,  // [B,S,C]
                      const _Float16* __restrict__ km,  // [B,S,C]
                      const _Float16* __restrict__ vt,  // [B,h,d,S]
                      float* __restrict__ attn,         // [B,H,S,S]
                      _Float16* __restrict__ ctx)       // [B,S,C] f16
{
  __shared__ __attribute__((aligned(16))) float sc[16 * SEQ];  // 64 KB
  _Float16* sc16 = (_Float16*)sc;       // f16 prob alias: bytes [0, 32K)
  float* red = sc + 8 * SEQ;            // free f32 slice:  bytes [32K, 36K)

  const int q0   = blockIdx.x * 16;
  const int h    = blockIdx.y;
  const int b    = blockIdx.z;
  const int tid  = threadIdx.x;
  const int wave = tid >> 5;
  const int lane = tid & 31;
  const int n    = lane & 15;
  const int hi   = lane >> 4;

  const _Float16* qb  = qm + (size_t)b * SEQ * HIDDEN + h * HDIM;
  const _Float16* kb  = km + (size_t)b * SEQ * HIDDEN + h * HDIM;
  const _Float16* vtb = vt + (size_t)(b * HEADS + h) * HDIM * SEQ;  // 128 KB

  // ---- prefetch the V block into L2 while we do QK^T + softmax ----
#pragma unroll
  for (int q = 0; q < 2; ++q)
    __builtin_prefetch(vtb + (size_t)tid * 256 + q * 128, 0, 1);

  // ---- Phase A: QK^T, double-buffered K fragments ----
  const size_t qrow = (size_t)(q0 + n) * HIDDEN;
  const v16h aq0 = frag_f16(qb + qrow, hi);
  const v16h aq1 = frag_f16(qb + qrow + 32, hi);

  {
    const _Float16* krow = kb + (size_t)(wave * 16 + n) * HIDDEN;
    v16h nb0 = frag_f16(krow, hi);
    v16h nb1 = frag_f16(krow + 32, hi);
    for (int kt = wave; kt < SEQ / 16; kt += 8) {
      const int ktn = (kt + 8) & 63;                    // wrapped: branchless
      const _Float16* krn = kb + (size_t)(ktn * 16 + n) * HIDDEN;
      const v16h b0 = nb0, b1 = nb1;
      nb0 = frag_f16(krn, hi);                          // next tile in flight
      nb1 = frag_f16(krn + 32, hi);
      v8f acc = {};
      acc = __builtin_amdgcn_wmma_f32_16x16x32_f16(false, aq0, false, b0,
                                                   (short)0, acc, false, false);
      acc = __builtin_amdgcn_wmma_f32_16x16x32_f16(false, aq1, false, b1,
                                                   (short)0, acc, false, false);
#pragma unroll
      for (int r = 0; r < 8; ++r)
        sc[(r + (hi << 3)) * SEQ + kt * 16 + n] = acc[r] * 0.125f;
    }
  }
  __syncthreads();

  // ---- Phase B: softmax, one row per half-wave ----
  {
    const int row = wave * 2 + hi;
    float vals[64];
    float pm = -3.4e38f;
#pragma unroll
    for (int j = 0; j < 64; ++j) {
      vals[j] = sc[row * SEQ + n + 16 * j];
      pm = fmaxf(pm, vals[j]);
    }
#pragma unroll
    for (int mk = 1; mk < 16; mk <<= 1)
      pm = fmaxf(pm, __shfl_xor(pm, mk, 32));
    float sum = 0.f;
#pragma unroll
    for (int j = 0; j < 64; ++j) {
      vals[j] = __expf(vals[j] - pm);
      sum += vals[j];
    }
#pragma unroll
    for (int mk = 1; mk < 16; mk <<= 1)
      sum += __shfl_xor(sum, mk, 32);
    const float inv = 1.0f / sum;

    __syncthreads();  // all f32 reads of sc done before f16 overwrite

#pragma unroll
    for (int j = 0; j < 64; ++j)
      sc16[row * SEQ + n + 16 * j] = (_Float16)(vals[j] * inv);
  }
  __syncthreads();

  // ---- Phase B2: stream attn block to HBM, 256 B contiguous per thread ----
  {
    float* ablk = attn + (size_t)((b * HEADS + h) * SEQ + q0) * SEQ;
    const _Float16* src = sc16 + tid * 64;              // flat row*1024+col
#pragma unroll
    for (int q = 0; q < 8; ++q) {
      const v8h hsrc = *(const v8h*)(src + 8 * q);      // ds b128
      v4f f0, f1;
#pragma unroll
      for (int i = 0; i < 4; ++i) { f0[i] = (float)hsrc[i]; f1[i] = (float)hsrc[i + 4]; }
      __builtin_nontemporal_store(f0, (v4f*)(ablk + (size_t)tid * 64 + 8 * q));
      __builtin_nontemporal_store(f1, (v4f*)(ablk + (size_t)tid * 64 + 8 * q + 4));
    }
  }

  // ---- Phase C: ctx = P @ V; 8 waves = 4 d-slices x 2 K-halves ----
  {
    const int ntile = wave & 3;
    const int khalf = wave >> 2;
    const int base  = khalf * (SEQ / 2);
    const _Float16* ps   = sc16 + (size_t)n * SEQ;
    const _Float16* vrow = vtb + (size_t)(ntile * 16 + n) * SEQ;

    v8f acc = {};
    v16h na = frag_f16(ps + base, hi);
    v16h nb = frag_f16(vrow + base, hi);
    for (int rel = 0; rel < SEQ / 2; rel += 32) {
      const int nxt = base + ((rel + 32) & (SEQ / 2 - 1));  // wrapped
      const v16h a = na, bv2 = nb;
      na = frag_f16(ps + nxt, hi);
      nb = frag_f16(vrow + nxt, hi);
      acc = __builtin_amdgcn_wmma_f32_16x16x32_f16(false, a, false, bv2,
                                                   (short)0, acc, false, false);
    }

    if (khalf == 1) {
#pragma unroll
      for (int r = 0; r < 8; ++r) red[(ntile * 32 + lane) * 8 + r] = acc[r];
    }
    __syncthreads();
    if (khalf == 0) {
#pragma unroll
      for (int r = 0; r < 8; ++r) acc[r] += red[(ntile * 32 + lane) * 8 + r];
      _Float16* cb = ctx + (size_t)b * SEQ * HIDDEN + h * HDIM + ntile * 16;
#pragma unroll
      for (int r = 0; r < 8; ++r)
        cb[(size_t)(q0 + r + 8 * hi) * HIDDEN + n] = (_Float16)acc[r];
    }
  }
}

// ---------------------------------------------------------------------------
// Output projection: out[b][o][s] = ctx[b][s][:] . Wp[o][:] + bp[o]
// One wave per 16x16 tile; all loads/stores are b128 vectors.
// ---------------------------------------------------------------------------
__global__ __launch_bounds__(32)
void out_proj(const _Float16* __restrict__ ctx,        // [B,S,C] f16
              const float* __restrict__ Wp,            // [C,C]
              const float* __restrict__ bp,            // [C]
              float* __restrict__ out)                 // [B,C,S]
{
  const int lane = threadIdx.x;
  const int m  = lane & 15;
  const int hi = lane >> 4;
  const int s0 = blockIdx.x * 16;
  const int n0 = blockIdx.y * 16;
  const int b  = blockIdx.z;
  const _Float16* cb = ctx + (size_t)b * SEQ * HIDDEN + (size_t)(s0 + m) * HIDDEN;

  v8f acc = {};
  for (int k0 = 0; k0 < HIDDEN; k0 += 32) {
    const v16h af = frag_f16(cb + k0, hi);
    const v16h bf = frag_f32(Wp + (size_t)(n0 + m) * HIDDEN + k0, hi);
    acc = __builtin_amdgcn_wmma_f32_16x16x32_f16(false, af, false, bf,
                                                 (short)0, acc, false, false);
  }
  const float bv = bp[n0 + m];
  float* orow = out + (size_t)b * HIDDEN * SEQ + (size_t)(n0 + m) * SEQ + s0 + 8 * hi;
  v4f o0, o1;
#pragma unroll
  for (int r = 0; r < 4; ++r) { o0[r] = acc[r] + bv; o1[r] = acc[r + 4] + bv; }
  *(v4f*)orow = o0;
  *(v4f*)(orow + 4) = o1;
}

// ---------------------------------------------------------------------------
extern "C" void kernel_launch(void* const* d_in, const int* in_sizes, int n_in,
                              void* d_out, int out_size, void* d_ws, size_t ws_size,
                              hipStream_t stream) {
  (void)in_sizes; (void)n_in; (void)out_size; (void)ws_size;

  const float* x  = (const float*)d_in[0];
  const float* Wq = (const float*)d_in[1];
  const float* bq = (const float*)d_in[2];
  const float* Wk = (const float*)d_in[3];
  const float* bk = (const float*)d_in[4];
  const float* Wv = (const float*)d_in[5];
  const float* bv = (const float*)d_in[6];
  const float* Wp = (const float*)d_in[7];
  const float* bp = (const float*)d_in[8];

  float* out  = (float*)d_out;                          // [16,256,32,32]
  float* attn = out + (size_t)BATCH * HIDDEN * SEQ;     // [16,4,1024,1024]

  const size_t mat = (size_t)BATCH * SEQ * HIDDEN;      // 4,194,304 elems
  _Float16* qm = (_Float16*)d_ws;                       // [B,S,C]
  _Float16* km = qm + mat;                              // [B,S,C]
  _Float16* vt = km + mat;                              // [B,h,d,S] (transposed)
  _Float16* cm = vt + mat;                              // [B,S,C]

  dim3 gp(SEQ / 16, 1, BATCH);                          // 256-thread proj blocks
  proj_kernel<0><<<gp, 256, 0, stream>>>(x, Wq, bq, qm);
  proj_kernel<0><<<gp, 256, 0, stream>>>(x, Wk, bk, km);
  proj_kernel<1><<<gp, 256, 0, stream>>>(x, Wv, bv, vt);

  dim3 ga(SEQ / 16, HEADS, BATCH);                      // (qtile, head, batch)
  attention_kernel<<<ga, 256, 0, stream>>>(qm, km, vt, attn, cm);

  dim3 go(SEQ / 16, HIDDEN / 16, BATCH);                // 1-wave tiles
  out_proj<<<go, 32, 0, stream>>>(cm, Wp, bp, out);
}
